// TemporalSelfRetention_56719338111656
// MI455X (gfx1250) — matrix operand order repaired and checked by reading, not compiled, not run on hardware
//
#include <hip/hip_runtime.h>
#include <hip/hip_bf16.h>
#include <math.h>

#define BB 2
#define TT 12
#define NN 207
#define DM 64
#define HH 4
#define DK 16
#define DH 256
#define TOK (BB*TT*NN)        // 4968 tokens
#define TOKP 4976             // padded to multiple of 16
#define NTILE (TOKP/16)       // 311
#define JHK 192               // 3*H*DK output units per token
#define W2COLS (JHK*DM)       // 12288

typedef __attribute__((ext_vector_type(16))) _Float16 v16h;
typedef __attribute__((ext_vector_type(8)))  _Float16 v8h;
typedef __attribute__((ext_vector_type(8)))  float    v8f;

// ---------------------------------------------------------------------------
// Build RetNet decay matrix D[h][t][s] (masked, column-sqrt-sum normalized)
// ---------------------------------------------------------------------------
__global__ void k_dmat(float* __restrict__ Dm) {
    int tid = threadIdx.x;
    if (tid < HH * TT) {
        int h = tid / TT, n = tid % TT;
        float r = 1.f - exp2f(-5.f - (float)h);
        float row[TT];
        float sum = 0.f, v = 1.f;
        for (int m = n; m >= 0; --m) { row[m] = v; sum += v; v *= r; }
        for (int m = n + 1; m < TT; ++m) row[m] = 0.f;
        float inv = rsqrtf(sum);
        for (int m = 0; m < TT; ++m) Dm[h*TT*TT + n*TT + m] = row[m] * inv;
    }
}

// ---------------------------------------------------------------------------
// Transpose ml_w2 [DH, W2COLS] f32 -> w2t [W2COLS, DH] f16 (LDS tile transpose)
// Gives contiguous 32B WMMA B-fragments along the K(=e) axis.
// ---------------------------------------------------------------------------
__global__ void k_w2t(const float* __restrict__ w2, _Float16* __restrict__ w2t) {
    __shared__ float tile[32][33];
    int tx = threadIdx.x, ty = threadIdx.y;          // 32 x 8
    int ebase = blockIdx.y * 32, cbase = blockIdx.x * 32;
#pragma unroll
    for (int i = 0; i < 4; ++i)
        tile[ty + i*8][tx] = w2[(size_t)(ebase + ty + i*8) * W2COLS + cbase + tx];
    __syncthreads();
#pragma unroll
    for (int i = 0; i < 4; ++i)
        w2t[(size_t)(cbase + ty + i*8) * DH + ebase + tx] = (_Float16)tile[tx][ty + i*8];
}

// ---------------------------------------------------------------------------
// MetaLearner hidden layer: hid[tok][e] = relu(c_in[tok]·ml_w1[:,e] + b1[e]) (f16)
// Pad rows [TOK, TOKP) with zeros so the WMMA kernel can load unguarded.
// ---------------------------------------------------------------------------
__global__ void k_hid(const float* __restrict__ cin, const float* __restrict__ w1,
                      const float* __restrict__ b1, _Float16* __restrict__ hid) {
    int tok = blockIdx.x, e = threadIdx.x;
    __shared__ float xs[DM];
    if (threadIdx.x < DM)
        xs[threadIdx.x] = (tok < TOK) ? cin[(size_t)tok*DM + threadIdx.x] : 0.f;
    __syncthreads();
    float s = b1[e];
#pragma unroll 8
    for (int d = 0; d < DM; ++d) s += xs[d] * w1[(size_t)d*DH + e];
    s = fmaxf(s, 0.f);
    hid[(size_t)tok*DH + e] = (tok < TOK) ? (_Float16)s : (_Float16)0.f;
}

// ---------------------------------------------------------------------------
// Dominant kernel: fused  W = hid @ ml_w2  (never materialized) with the
// per-token contraction  qkv[t, jhk] = sum_d (W[t, jhk*64+d] + b2) * x[t,d].
// One block = 16 tokens, 4 waves; wave w owns jhk = w*48 .. w*48+47.
// Per jhk: 16x64 slice via 4 N-tiles x 8 K-steps of v_wmma_f32_16x16x32_f16.
// ---------------------------------------------------------------------------
__global__ __launch_bounds__(128) void k_qkv(const _Float16* __restrict__ hid,
        const _Float16* __restrict__ w2t, const float* __restrict__ b2,
        const float* __restrict__ xin, float* __restrict__ qkv) {
    __shared__ alignas(16) _Float16 sA[16][264];   // hid tile, +8 halves pad
    __shared__ float sX[16][DM];                   // inputs tile
    const int tid  = threadIdx.x;
    const int lane = tid & 31, wave = tid >> 5;
    const int hf   = lane >> 4, nl = lane & 15;    // lane-half, lane-in-half
    const int tokBase = blockIdx.x * 16;

    // Stage A (f16 hid rows, zero-padded globally) and X (f32 inputs, guarded)
    for (int r = 0; r < 16; ++r) {
        unsigned int u = *(const unsigned int*)&hid[(size_t)(tokBase + r)*DH + tid*2];
        *(unsigned int*)&sA[r][tid*2] = u;
    }
#pragma unroll
    for (int i = 0; i < 8; ++i) {
        int flat = i*128 + tid, r = flat >> 6, c = flat & 63;
        int tok = tokBase + r;
        sX[r][c] = (tok < TOK) ? xin[(size_t)tok*DM + c] : 0.f;
    }
    __syncthreads();

    // Preload the 8 A-fragments: row m = lane%16,
    // halves 0..15 = K[koff..koff+15] (contiguous, matching B),
    // koff = 16*(lane/16) + 32*kstep.
    v16h afrag[8];
#pragma unroll
    for (int ks = 0; ks < 8; ++ks) {
        int koff = hf*16 + ks*32;
        v8h lo = *(const v8h*)&sA[nl][koff];
        v8h hi = *(const v8h*)&sA[nl][koff + 8];
        afrag[ks] = __builtin_shufflevector(lo, hi,
                        0,1,2,3,4,5,6,7,8,9,10,11,12,13,14,15);
    }

    for (int i = 0; i < 48; ++i) {
        const int jhk = wave*48 + i;
        // prefetch next jhk's B columns into cache (global_prefetch_b8)
        __builtin_prefetch(w2t + (size_t)((jhk + 1)*64 + nl)*DH + hf*16, 0, 3);

        float b2v[4];
#pragma unroll
        for (int nt = 0; nt < 4; ++nt) b2v[nt] = b2[jhk*64 + nt*16 + nl];

        v8f accs[4];
#pragma unroll
        for (int nt = 0; nt < 4; ++nt) {
            // B-fragment (16-bit B 32x16, contiguous k order as in A): col n = lane%16,
            // K = {0..15} + 16*(lane/16) + 32*kstep -> contiguous along e in w2t.
            const _Float16* bcol =
                w2t + (size_t)(jhk*64 + nt*16 + nl)*DH + hf*16;
            v8f acc = {};
#pragma unroll
            for (int ks = 0; ks < 8; ++ks) {
                v16h bf = *(const v16h*)(bcol + ks*32);
                acc = __builtin_amdgcn_wmma_f32_16x16x32_f16(
                        false, afrag[ks], false, bf, (short)0, acc, false, false);
            }
            accs[nt] = acc;
        }

        // Epilogue: D-layout element (m = v + 8*hf, d = nt*16 + nl);
        // add bias, multiply by x[m][d], reduce d across the 16-lane half.
#pragma unroll
        for (int v = 0; v < 8; ++v) {
            int m = v + hf*8;
            float s = 0.f;
#pragma unroll
            for (int nt = 0; nt < 4; ++nt)
                s += (accs[nt][v] + b2v[nt]) * sX[m][nt*16 + nl];
            s += __shfl_xor(s, 1, 32);
            s += __shfl_xor(s, 2, 32);
            s += __shfl_xor(s, 4, 32);
            s += __shfl_xor(s, 8, 32);
            if (nl == 0) {
                int tok = tokBase + m;
                if (tok < TOK) qkv[(size_t)tok*JHK + jhk] = s;
            }
        }
    }
}

// ---------------------------------------------------------------------------
// Retention per (b, n, h): scores -> decay mask -> abs-sum norm -> V,
// GroupNorm over dk (nan->0), 3*tanh. Block = 192 threads = (t, k/s).
// ---------------------------------------------------------------------------
__global__ __launch_bounds__(192) void k_ret(const float* __restrict__ qkv,
        const float* __restrict__ Dm, float* __restrict__ x2) {
    const int h  = blockIdx.x & 3;
    const int nn = (blockIdx.x >> 2) % NN;
    const int b  = blockIdx.x / (4*NN);
    const int tid = threadIdx.x;
    const int t = tid >> 4, u = tid & 15;
    __shared__ float Qs[TT][DK], Ks[TT][DK], Vs[TT][DK], ret[TT][TT], den[TT];

    const size_t tok = (size_t)(b*TT + t)*NN + nn;
    const float* base = qkv + tok*JHK + h*DK;
    Qs[t][u] = base[u];
    Ks[t][u] = base[64 + u];
    Vs[t][u] = base[128 + u];
    __syncthreads();

    if (u < TT) {
        int s_ = u;
        float dp = 0.f;
#pragma unroll
        for (int k = 0; k < DK; ++k) dp += Qs[t][k] * Ks[s_][k];
        float ct = cosf((float)(t + 1)),  st = sinf((float)(t + 1));
        float cs = cosf((float)(s_ + 1)), ss = sinf((float)(s_ + 1));
        ret[t][s_] = dp * 0.25f * (ct*cs + st*ss) * Dm[h*TT*TT + t*TT + s_];
    }
    __syncthreads();
    if (u == 0) {
        float sm = 0.f;
        for (int s_ = 0; s_ < TT; ++s_) sm += ret[t][s_];
        den[t] = fmaxf(fabsf(sm), 1.0f);
    }
    __syncthreads();

    float o = 0.f;
    for (int s_ = 0; s_ < TT; ++s_) o += ret[t][s_] * Vs[s_][u];
    o /= den[t];

    // GroupNorm over the 16 k-lanes (aligned 16-lane groups inside wave32)
    float s1 = o;
    s1 += __shfl_xor(s1, 1, 32); s1 += __shfl_xor(s1, 2, 32);
    s1 += __shfl_xor(s1, 4, 32); s1 += __shfl_xor(s1, 8, 32);
    float mu = s1 * (1.f/16.f);
    float d  = o - mu;
    float s2 = d * d;
    s2 += __shfl_xor(s2, 1, 32); s2 += __shfl_xor(s2, 2, 32);
    s2 += __shfl_xor(s2, 4, 32); s2 += __shfl_xor(s2, 8, 32);
    float var = s2 * (1.f/16.f);
    float val = d / sqrtf(var);
    if (!(val == val)) val = 0.f;          // nan -> 0
    x2[tok*DM + h*DK + u] = 3.f * tanhf(val);
}

// ---------------------------------------------------------------------------
// GDC (softmax-over-heads gate) + Swish gate + output projection + residual LN
// Block = 1 token, 64 threads = output channel m.
// ---------------------------------------------------------------------------
__global__ __launch_bounds__(64) void k_out(const float* __restrict__ xin,
        const float* __restrict__ x2, const float* __restrict__ g1,
        const float* __restrict__ g2, const float* __restrict__ wgw,
        const float* __restrict__ wgb, const float* __restrict__ wow,
        const float* __restrict__ wob, const float* __restrict__ lng,
        const float* __restrict__ lnb, float* __restrict__ out) {
    const int tok = blockIdx.x, m = threadIdx.x;
    __shared__ float xs[DM], x2s[DM], zs[DM], red[4];
    xs[m]  = xin[(size_t)tok*DM + m];
    x2s[m] = x2[(size_t)tok*DM + m];
    __syncthreads();

    float a[HH], sr[HH];
#pragma unroll
    for (int h = 0; h < HH; ++h) {
        float aa = 0.f, ssv = 0.f;
#pragma unroll
        for (int k = 0; k < DK; ++k) {
            float xv = x2s[h*DK + k];
            aa  += xv * g1[(h*DK + k)*DM + m];
            ssv += xv * g2[(h*DK + k)*DM + m];
        }
        a[h] = aa; sr[h] = fmaxf(ssv, 0.f);
    }
    float mx = fmaxf(fmaxf(sr[0], sr[1]), fmaxf(sr[2], sr[3]));
    float e0 = expf(sr[0]-mx), e1 = expf(sr[1]-mx),
          e2 = expf(sr[2]-mx), e3 = expf(sr[3]-mx);
    float g = (a[0]*e0 + a[1]*e1 + a[2]*e2 + a[3]*e3) / (e0+e1+e2+e3);

    float z = wgb[m];
    for (int d = 0; d < DM; ++d) z += xs[d] * wgw[d*DM + m];
    z *= g;
    z = z / (1.f + expf(-z));              // z * sigmoid(z)
    zs[m] = z;
    __syncthreads();

    float o = wob[m];
    for (int d = 0; d < DM; ++d) o += zs[d] * wow[d*DM + m];
    float y = o + xs[m];

    // LayerNorm over 64 channels (2 waves): shfl within wave, LDS across
    float s1 = y, s2 = y*y;
    for (int msk = 1; msk < 32; msk <<= 1) {
        s1 += __shfl_xor(s1, msk, 32);
        s2 += __shfl_xor(s2, msk, 32);
    }
    int lane = m & 31, wv = m >> 5;
    if (lane == 0) { red[wv] = s1; red[2 + wv] = s2; }
    __syncthreads();
    float S1 = red[0] + red[1], S2 = red[2] + red[3];
    float mu  = S1 * (1.f/64.f);
    float var = S2 * (1.f/64.f) - mu*mu;
    out[(size_t)tok*DM + m] = (y - mu) * rsqrtf(var + 1e-5f) * lng[m] + lnb[m];
}

// ---------------------------------------------------------------------------
extern "C" void kernel_launch(void* const* d_in, const int* in_sizes, int n_in,
                              void* d_out, int out_size, void* d_ws, size_t ws_size,
                              hipStream_t stream) {
    (void)in_sizes; (void)n_in; (void)out_size; (void)ws_size;
    const float* inp  = (const float*)d_in[0];
    const float* cin  = (const float*)d_in[1];
    const float* mlw1 = (const float*)d_in[2];
    const float* mlb1 = (const float*)d_in[3];
    const float* mlw2 = (const float*)d_in[4];
    const float* mlb2 = (const float*)d_in[5];
    const float* g1   = (const float*)d_in[6];
    const float* g2   = (const float*)d_in[7];
    const float* wgw  = (const float*)d_in[8];
    const float* wgb  = (const float*)d_in[9];
    const float* wow  = (const float*)d_in[10];
    const float* wob  = (const float*)d_in[11];
    const float* lng  = (const float*)d_in[12];
    const float* lnb  = (const float*)d_in[13];
    float* out = (float*)d_out;

    char* ws = (char*)d_ws;
    size_t o = 0;
    _Float16* w2t = (_Float16*)(ws + o);  o += (size_t)W2COLS * DH * 2;  // 6.29 MB
    _Float16* hid = (_Float16*)(ws + o);  o += (size_t)TOKP * DH * 2;    // 2.55 MB
    float*    qkv = (float*)(ws + o);     o += (size_t)TOK * JHK * 4;    // 3.82 MB
    float*    x2  = (float*)(ws + o);     o += (size_t)TOK * DM * 4;     // 1.27 MB
    float*    Dm  = (float*)(ws + o);     o += (size_t)HH * TT * TT * 4;

    hipLaunchKernelGGL(k_dmat, dim3(1), dim3(64), 0, stream, Dm);
    hipLaunchKernelGGL(k_w2t, dim3(W2COLS/32, DH/32), dim3(32, 8), 0, stream, mlw2, w2t);
    hipLaunchKernelGGL(k_hid, dim3(TOKP), dim3(DH), 0, stream, cin, mlw1, mlb1, hid);
    hipLaunchKernelGGL(k_qkv, dim3(NTILE), dim3(128), 0, stream, hid, w2t, mlb2, inp, qkv);
    hipLaunchKernelGGL(k_ret, dim3(BB*NN*HH), dim3(192), 0, stream, qkv, Dm, x2);
    hipLaunchKernelGGL(k_out, dim3(TOK), dim3(64), 0, stream,
                       inp, x2, g1, g2, wgw, wgb, wow, wob, lng, lnb, out);
}
